// CausalSelfAttention_60060822667899
// MI455X (gfx1250) — matrix operand order repaired and checked by reading, not compiled, not run on hardware
//
#include <hip/hip_runtime.h>
#include <hip/hip_bf16.h>
#include <math.h>

typedef _Float16 half_t;
typedef __attribute__((ext_vector_type(16))) _Float16 v16h;
typedef __attribute__((ext_vector_type(8)))  _Float16 v8h;
typedef __attribute__((ext_vector_type(8)))  float    v8f;
typedef __attribute__((ext_vector_type(4)))  float    fl4;

#define WMMA_F16(a, b, c) \
  __builtin_amdgcn_wmma_f32_16x16x32_f16(false, (a), false, (b), (short)0, (c), false, false)

static __device__ __forceinline__ v16h cat16(v8h lo, v8h hi) {
  v16h r;
#pragma unroll
  for (int i = 0; i < 8; i++) { r[i] = lo[i]; r[i + 8] = hi[i]; }
  return r;
}
static __device__ __forceinline__ v8f zero8() {
  v8f z;
#pragma unroll
  for (int i = 0; i < 8; i++) z[i] = 0.0f;
  return z;
}
static __device__ __forceinline__ v16h cvt16(const fl4* q) {
  v16h t;
#pragma unroll
  for (int j = 0; j < 4; j++) {
    t[j]      = (half_t)q[0][j];
    t[j + 4]  = (half_t)q[1][j];
    t[j + 8]  = (half_t)q[2][j];
    t[j + 12] = (half_t)q[3][j];
  }
  return t;
}

// ---------------------------------------------------------------------------
// Kernel 0: transpose + convert  dst[N][K] (f16) = src[K][N] (f32).
// ---------------------------------------------------------------------------
__global__ __launch_bounds__(256) void k0_tcvt(
    const float* __restrict__ src, half_t* __restrict__ dst, int R, int C) {
  __shared__ half_t tile[64 * 66];
  const int r0 = blockIdx.x * 64;
  const int c0 = blockIdx.y * 64;
  const int tid = threadIdx.x;
#pragma unroll 4
  for (int it = 0; it < 16; ++it) {
    const int idx = it * 256 + tid;
    const int cc = idx & 63, rr = idx >> 6;
    tile[rr * 66 + cc] = (half_t)src[(size_t)(r0 + rr) * C + (c0 + cc)];
  }
  __syncthreads();
#pragma unroll 4
  for (int it = 0; it < 16; ++it) {
    const int idx = it * 256 + tid;
    const int rr = idx & 63, cc = idx >> 6;
    dst[(size_t)(c0 + cc) * R + (r0 + rr)] = tile[rr * 66 + cc];
  }
}

// ---------------------------------------------------------------------------
// Kernel 1: qkv = x @ w_attn (M=4096,N=6144,K=2048), fused RoPE epilogue.
// 8 waves/block, 32x64 tile per wave, LDS-free (L2-resident operands).
// Software-pipelined: kb+1 raw loads issued before kb fragments consumed.
// ---------------------------------------------------------------------------
__global__ __launch_bounds__(256) void k1_qkv_rope(
    const float* __restrict__ x, const half_t* __restrict__ wT,
    const int* __restrict__ startp,
    half_t* __restrict__ qws, half_t* __restrict__ kws,
    half_t* __restrict__ vtws) {
  const int tid  = threadIdx.x;
  const int wave = tid >> 5;
  const int lane = tid & 31;
  const int hfl  = lane >> 4;
  const int ln   = lane & 15;
  const int m0 = blockIdx.x * 32;
  const int nw = blockIdx.y * 512 + wave * 64;

  const float* xrow[2];
#pragma unroll
  for (int am = 0; am < 2; ++am)
    xrow[am] = x + (size_t)(m0 + am * 16 + ln) * 2048 + 8 * hfl;
  const half_t* wrow[4];
#pragma unroll
  for (int a = 0; a < 4; ++a)
    wrow[a] = wT + (size_t)(nw + a * 16 + ln) * 2048 + 8 * hfl;

  v8f acc[2][4];
#pragma unroll
  for (int i = 0; i < 2; i++)
#pragma unroll
    for (int j = 0; j < 4; j++) acc[i][j] = zero8();

  fl4 xc[2][4];
  v8h wc[4][2];
#pragma unroll
  for (int am = 0; am < 2; ++am) {
    xc[am][0] = *(const fl4*)(xrow[am] + 0);
    xc[am][1] = *(const fl4*)(xrow[am] + 4);
    xc[am][2] = *(const fl4*)(xrow[am] + 16);
    xc[am][3] = *(const fl4*)(xrow[am] + 20);
  }
#pragma unroll
  for (int a = 0; a < 4; ++a) {
    wc[a][0] = *(const v8h*)(wrow[a]);
    wc[a][1] = *(const v8h*)(wrow[a] + 16);
  }

  for (int kb = 0; kb < 64; ++kb) {
    fl4 xn[2][4];
    v8h wn[4][2];
    if (kb < 63) {
      const int off = (kb + 1) * 32;
#pragma unroll
      for (int am = 0; am < 2; ++am) {
        xn[am][0] = *(const fl4*)(xrow[am] + off);
        xn[am][1] = *(const fl4*)(xrow[am] + off + 4);
        xn[am][2] = *(const fl4*)(xrow[am] + off + 16);
        xn[am][3] = *(const fl4*)(xrow[am] + off + 20);
      }
#pragma unroll
      for (int a = 0; a < 4; ++a) {
        wn[a][0] = *(const v8h*)(wrow[a] + off);
        wn[a][1] = *(const v8h*)(wrow[a] + off + 16);
        if (kb < 62) __builtin_prefetch((const void*)(wrow[a] + off + 32), 0, 1);
      }
    }
    v16h af[2];
#pragma unroll
    for (int am = 0; am < 2; ++am) af[am] = cvt16(xc[am]);
#pragma unroll
    for (int a = 0; a < 4; ++a) {
      const v16h bf = cat16(wc[a][0], wc[a][1]);
#pragma unroll
      for (int am = 0; am < 2; ++am)
        acc[am][a] = WMMA_F16(af[am], bf, acc[am][a]);
    }
    if (kb < 63) {
#pragma unroll
      for (int am = 0; am < 2; ++am)
#pragma unroll
        for (int j = 0; j < 4; j++) xc[am][j] = xn[am][j];
#pragma unroll
      for (int a = 0; a < 4; ++a) {
        wc[a][0] = wn[a][0];
        wc[a][1] = wn[a][1];
      }
    }
  }

  const int start = *startp;
  const float lg = logf(10000.0f) / 64.0f;
#pragma unroll
  for (int am = 0; am < 2; ++am) {
#pragma unroll
    for (int a = 0; a < 4; a++) {
      const v8f c = acc[am][a];
      const int n   = nw + a * 16 + ln;
      const int seg = n >> 11;
      const int cc  = n & 2047;
      const int h   = cc >> 7;
      const int d   = cc & 127;
      if (seg < 2) {
        half_t* dst = (seg == 0) ? qws : kws;
        const float invf = __expf(-(float)(d >> 1) * lg);
        const bool odd = (d & 1);
#pragma unroll
        for (int r = 0; r < 8; r++) {
          const int m = m0 + am * 16 + hfl * 8 + r;
          const int b = m >> 11;
          const int t = m & 2047;
          const float own = c[r];
          const float oth = __shfl_xor(own, 1, 32);
          float sn, cs;
          __sincosf((float)(t + start) * invf, &sn, &cs);
          const float rv = odd ? (oth * sn + own * cs) : (own * cs - oth * sn);
          dst[((size_t)(b * 16 + h) * 2048 + t) * 128 + d] = (half_t)rv;
        }
      } else {
#pragma unroll
        for (int r = 0; r < 8; r++) {
          const int m = m0 + am * 16 + hfl * 8 + r;
          const int b = m >> 11;
          const int t = m & 2047;
          vtws[((size_t)(b * 16 + h) * 128 + d) * 2048 + t] = (half_t)c[r];
        }
      }
    }
  }
}

// ---------------------------------------------------------------------------
// Kernel 2: flash attention, one wave per (b,h,16-query tile).
// ---------------------------------------------------------------------------
__global__ __launch_bounds__(32) void k2_attn(
    const half_t* __restrict__ qws, const half_t* __restrict__ kws,
    const half_t* __restrict__ vtws, half_t* __restrict__ yws) {
  __shared__ float  sld[16 * 33];
  __shared__ half_t pld[16 * 40];
  __shared__ float  fld[16];
  const int lane = threadIdx.x;
  const int hfl = lane >> 4, ln = lane & 15;
  const int qt = blockIdx.x & 127;
  const int bh = blockIdx.x >> 7;
  const int q0 = qt * 16;
  const half_t* qb  = qws  + (size_t)bh * 2048 * 128;
  const half_t* kbp = kws  + (size_t)bh * 2048 * 128;
  const half_t* vb  = vtws + (size_t)bh * 128 * 2048;

  v16h qf[4];
  {
    const half_t* qrow = qb + (size_t)(q0 + ln) * 128;
#pragma unroll
    for (int c2 = 0; c2 < 4; c2++) {
      v8h lo = *(const v8h*)(qrow + 32 * c2 + 16 * hfl);
      v8h hi = *(const v8h*)(qrow + 32 * c2 + 16 * hfl + 8);
      qf[c2] = cat16(lo, hi);
    }
  }
  v8f o[8];
#pragma unroll
  for (int i = 0; i < 8; i++) o[i] = zero8();
  float mrun = -1e30f, lrun = 0.0f;

  const int nkb = (q0 + 16 + 31) >> 5;
  for (int kb = 0; kb < nkb; ++kb) {
    const int k0 = kb * 32;
    v8f s0 = zero8(), s1 = zero8();
    const half_t* kr0 = kbp + (size_t)(k0 + ln) * 128;
    const half_t* kr1 = kbp + (size_t)(k0 + 16 + ln) * 128;
#pragma unroll
    for (int c2 = 0; c2 < 4; c2++) {
      v8h l0 = *(const v8h*)(kr0 + 32 * c2 + 16 * hfl);
      v8h h0 = *(const v8h*)(kr0 + 32 * c2 + 16 * hfl + 8);
      v8h l1 = *(const v8h*)(kr1 + 32 * c2 + 16 * hfl);
      v8h h1 = *(const v8h*)(kr1 + 32 * c2 + 16 * hfl + 8);
      s0 = WMMA_F16(qf[c2], cat16(l0, h0), s0);
      s1 = WMMA_F16(qf[c2], cat16(l1, h1), s1);
    }
    // Issue all V-fragment loads now; latency hidden behind softmax phase.
    v8h vlo[8], vhi[8];
#pragma unroll
    for (int dt = 0; dt < 8; ++dt) {
      const half_t* vr = vb + (size_t)(dt * 16 + ln) * 2048 + k0;
      vlo[dt] = *(const v8h*)(vr + 8 * hfl);
      vhi[dt] = *(const v8h*)(vr + 16 + 8 * hfl);
    }
    const float sc = 0.08838834764831845f;  // 1/sqrt(128)
#pragma unroll
    for (int r = 0; r < 8; r++) {
      const int ql = hfl * 8 + r;
      const int qi = q0 + ql;
      sld[ql * 33 + ln]      = (k0 + ln      <= qi) ? s0[r] * sc : -1e30f;
      sld[ql * 33 + 16 + ln] = (k0 + 16 + ln <= qi) ? s1[r] * sc : -1e30f;
    }
    asm volatile("s_wait_dscnt 0" ::: "memory");
    if (lane < 16) {
      float mx = -1e30f;
#pragma unroll
      for (int j = 0; j < 32; j++) mx = fmaxf(mx, sld[lane * 33 + j]);
      const float mnew = fmaxf(mrun, mx);
      const float resc = __expf(mrun - mnew);
      float sum = 0.0f;
#pragma unroll
      for (int j = 0; j < 32; j++) {
        const float p = __expf(sld[lane * 33 + j] - mnew);
        sum += p;
        pld[lane * 40 + j] = (half_t)p;
      }
      lrun = lrun * resc + sum;
      mrun = mnew;
      fld[lane] = resc;
    }
    asm volatile("s_wait_dscnt 0" ::: "memory");
    const float resc = fld[ln];
    v8h plo = *(const v8h*)&pld[ln * 40 + 8 * hfl];
    v8h phi = *(const v8h*)&pld[ln * 40 + 16 + 8 * hfl];
    const v16h pf = cat16(plo, phi);
#pragma unroll
    for (int dt = 0; dt < 8; ++dt) {
#pragma unroll
      for (int r = 0; r < 8; r++) o[dt][r] *= resc;
      o[dt] = WMMA_F16(cat16(vlo[dt], vhi[dt]), pf, o[dt]);
    }
  }
  if (lane < 16) fld[lane] = 1.0f / lrun;
  asm volatile("s_wait_dscnt 0" ::: "memory");
  const float inv = fld[ln];
  const int b = bh >> 4, h = bh & 15;
  const int t = q0 + ln;
  half_t* yrow = yws + ((size_t)(b * 2048 + t) * 2048) + h * 128;
#pragma unroll
  for (int dt = 0; dt < 8; ++dt) {
    v8h ov;
#pragma unroll
    for (int r = 0; r < 8; r++) ov[r] = (half_t)(o[dt][r] * inv);
    *(v8h*)(yrow + dt * 16 + 8 * hfl) = ov;
  }
}

// ---------------------------------------------------------------------------
// Kernel 3: out = y @ w_proj (M=4096,N=2048,K=2048), fp32 output.
// ---------------------------------------------------------------------------
__global__ __launch_bounds__(256) void k3_proj(
    const half_t* __restrict__ yws, const half_t* __restrict__ wT,
    float* __restrict__ out) {
  const int tid  = threadIdx.x;
  const int wave = tid >> 5;
  const int lane = tid & 31;
  const int hfl  = lane >> 4;
  const int ln   = lane & 15;
  const int m0 = blockIdx.x * 32;
  const int nw = blockIdx.y * 512 + wave * 64;

  const half_t* yrow[2];
#pragma unroll
  for (int am = 0; am < 2; ++am)
    yrow[am] = yws + (size_t)(m0 + am * 16 + ln) * 2048 + 8 * hfl;
  const half_t* wrow[4];
#pragma unroll
  for (int a = 0; a < 4; ++a)
    wrow[a] = wT + (size_t)(nw + a * 16 + ln) * 2048 + 8 * hfl;

  v8f acc[2][4];
#pragma unroll
  for (int i = 0; i < 2; i++)
#pragma unroll
    for (int j = 0; j < 4; j++) acc[i][j] = zero8();

  v8h yc[2][2], wc[4][2];
#pragma unroll
  for (int am = 0; am < 2; ++am) {
    yc[am][0] = *(const v8h*)(yrow[am]);
    yc[am][1] = *(const v8h*)(yrow[am] + 16);
  }
#pragma unroll
  for (int a = 0; a < 4; ++a) {
    wc[a][0] = *(const v8h*)(wrow[a]);
    wc[a][1] = *(const v8h*)(wrow[a] + 16);
  }

  for (int kb = 0; kb < 64; ++kb) {
    v8h yn[2][2], wn[4][2];
    if (kb < 63) {
      const int off = (kb + 1) * 32;
#pragma unroll
      for (int am = 0; am < 2; ++am) {
        yn[am][0] = *(const v8h*)(yrow[am] + off);
        yn[am][1] = *(const v8h*)(yrow[am] + off + 16);
      }
#pragma unroll
      for (int a = 0; a < 4; ++a) {
        wn[a][0] = *(const v8h*)(wrow[a] + off);
        wn[a][1] = *(const v8h*)(wrow[a] + off + 16);
        if (kb < 62) __builtin_prefetch((const void*)(wrow[a] + off + 32), 0, 1);
      }
    }
    v16h af[2];
#pragma unroll
    for (int am = 0; am < 2; ++am) af[am] = cat16(yc[am][0], yc[am][1]);
#pragma unroll
    for (int a = 0; a < 4; ++a) {
      const v16h bf = cat16(wc[a][0], wc[a][1]);
#pragma unroll
      for (int am = 0; am < 2; ++am)
        acc[am][a] = WMMA_F16(af[am], bf, acc[am][a]);
    }
    if (kb < 63) {
#pragma unroll
      for (int am = 0; am < 2; ++am) {
        yc[am][0] = yn[am][0];
        yc[am][1] = yn[am][1];
      }
#pragma unroll
      for (int a = 0; a < 4; ++a) {
        wc[a][0] = wn[a][0];
        wc[a][1] = wn[a][1];
      }
    }
  }

#pragma unroll
  for (int am = 0; am < 2; ++am) {
#pragma unroll
    for (int a = 0; a < 4; a++) {
      const int n = nw + a * 16 + ln;
#pragma unroll
      for (int r = 0; r < 8; r++) {
        const int m = m0 + am * 16 + hfl * 8 + r;
        out[(size_t)m * 2048 + n] = acc[am][a][r];
      }
    }
  }
}

// ---------------------------------------------------------------------------
extern "C" void kernel_launch(void* const* d_in, const int* in_sizes, int n_in,
                              void* d_out, int out_size, void* d_ws,
                              size_t ws_size, hipStream_t stream) {
  const float* x       = (const float*)d_in[0];  // [2,2048,2048]
  const float* w_attn  = (const float*)d_in[1];  // [2048,6144]
  const float* w_proj  = (const float*)d_in[2];  // [2048,2048]
  const int*   startp  = (const int*)d_in[3];    // scalar
  float*       out     = (float*)d_out;          // [2,2048,2048]

  const size_t QKV = (size_t)2 * 16 * 2048 * 128;  // 8,388,608 f16 elems
  half_t* qws  = (half_t*)d_ws;
  half_t* kws  = qws + QKV;
  half_t* vtws = kws + QKV;
  half_t* yws  = vtws + QKV;                        // [2,2048,2048] f16
  half_t* wTa  = yws + QKV;                         // [6144,2048] f16
  half_t* wTp  = wTa + (size_t)6144 * 2048;         // [2048,2048] f16
  // total workspace: 4*16MB + 24MB + 8MB = 96MB

  dim3 gta(32, 96);
  k0_tcvt<<<gta, 256, 0, stream>>>(w_attn, wTa, 2048, 6144);
  dim3 gtp(32, 32);
  k0_tcvt<<<gtp, 256, 0, stream>>>(w_proj, wTp, 2048, 2048);

  dim3 g1(128, 12);
  k1_qkv_rope<<<g1, 256, 0, stream>>>(x, wTa, startp, qws, kws, vtws);
  k2_attn<<<4096, 32, 0, stream>>>(qws, kws, vtws, yws);
  dim3 g3(128, 4);
  k3_proj<<<g3, 256, 0, stream>>>(yws, wTp, out);
}
